// Net_29583734734938
// MI455X (gfx1250) — matrix operand order repaired and checked by reading, not compiled, not run on hardware
//
#include <hip/hip_runtime.h>

// Criss-cross attention (CCNet) for B=8, C=64, H=W=128 on gfx1250.
// 3 kernels: WMMA QKV projection, row attention (partial online softmax),
// column attention + softmax merge + residual. All matmuls on v_wmma_f32_16x16x32_f16.

#define B_  8
#define C_  64
#define C8_ 8
#define H_  128
#define W_  128

typedef __attribute__((ext_vector_type(16))) _Float16 v16h;
typedef __attribute__((ext_vector_type(8)))  float    v8f;

#define QK_LD 40     // row stride (halves) for Q/K tiles in LDS (K padded to 32)
#define PS_LD 136    // row stride for P (exp scores), 128 cols + pad
#define VS_LD 136    // row stride for V tiles, 128 cols + pad
#define PS_ELEMS (H_ * PS_LD)
#define VS_ELEMS (C_ * VS_LD)
#define SMEM_HALVES (PS_ELEMS + VS_ELEMS)   // Q/K region aliases the P region

// QKV kernel LDS geometry
#define ALD   136    // A tile row stride (128 in-ch + pad)
#define WLD   136    // Wv row stride (128 + pad)
#define QWLD  72     // Wq/Wk row stride (64 + pad)
#define OLD   136    // output staging row stride (128 spatial + pad)
#define A_ELEMS  (128 * ALD)                 // 17408 halves
#define WV_ELEMS (C_ * WLD)                  // 8704
#define WQ_ELEMS (16 * QWLD)                 // 1152
#define QSM_HALVES (A_ELEMS + WV_ELEMS + 2 * WQ_ELEMS)

__device__ __forceinline__ v8f wmma_f16f32(v16h a, v16h b, v8f c) {
  // D = A(16x32 f16) * B(32x16 f16) + C(16x16 f32)
  return __builtin_amdgcn_wmma_f32_16x16x32_f16(false, a, false, b, (short)0, c,
                                                false, false);
}

__device__ __forceinline__ v8f vzero8() {
  v8f z;
#pragma unroll
  for (int e = 0; e < 8; ++e) z[e] = 0.0f;
  return z;
}

// A fragment (16x32 f16) from row-major LDS [M][K] storage.
// lane<16: rows M=row0+lane, K = k0 + {0..7, 16..23}
// lane>=16: same rows,       K = k0 + {8..15, 24..31}
__device__ __forceinline__ v16h load_fragA(const _Float16* src, int ld,
                                           int row0, int k0, int lane) {
  const _Float16* rp = src + (row0 + (lane & 15)) * ld + k0 + ((lane & 16) ? 8 : 0);
  v16h a;
#pragma unroll
  for (int e = 0; e < 16; ++e) a[e] = rp[(e & 7) + ((e & 8) << 1)];
  return a;
}

// B fragment (32x16 f16) from transposed ([N][K] row-major) LDS storage.
// lane<16: col N=n0+lane, K = k0 + {0..7, 16..23} ; lane>=16: K = k0 + {8..15, 24..31}
__device__ __forceinline__ v16h load_fragBT(const _Float16* src, int ld,
                                            int n0, int k0, int lane) {
  const _Float16* rp = src + (n0 + (lane & 15)) * ld + k0 + ((lane & 16) ? 8 : 0);
  v16h b;
#pragma unroll
  for (int e = 0; e < 16; ++e) b[e] = rp[(e & 7) + ((e & 8) << 1)];
  return b;
}

// ---------------------------------------------------------------------------
// Kernel 1: QKV projection as WMMA GEMM. Block = 8x16 pixel tile (128 spatial
// positions), 256 threads / 8 waves; wave m owns spatial rows 16m..16m+15.
// out[s, och] = sum_ich [x;y][s, ich] * W[och, ich] + b[och].
// Writes q,k,v in f16 in both [h][w] and transposed [w][h] layouts.
// ---------------------------------------------------------------------------
__global__ __launch_bounds__(256)
void cc_qkv_kernel(const float* __restrict__ x, const float* __restrict__ y,
                   const float* __restrict__ Wq, const float* __restrict__ bq,
                   const float* __restrict__ Wk, const float* __restrict__ bk,
                   const float* __restrict__ Wv, const float* __restrict__ bv,
                   _Float16* __restrict__ qbuf, _Float16* __restrict__ kbuf,
                   _Float16* __restrict__ qT,   _Float16* __restrict__ kT,
                   _Float16* __restrict__ vbuf, _Float16* __restrict__ vT)
{
  __shared__ alignas(16) _Float16 qsm[QSM_HALVES];
  _Float16* As   = qsm;                              // [128 s][ALD]
  _Float16* Wv_h = qsm + A_ELEMS;                    // [64 och][WLD]
  _Float16* Wq_h = qsm + A_ELEMS + WV_ELEMS;         // [16 och][QWLD], rows 8..15 zero
  _Float16* Wk_h = qsm + A_ELEMS + WV_ELEMS + WQ_ELEMS;
  // output staging aliases the A region (after a barrier)
  _Float16* Osv = qsm;                               // [64 ch][OLD]
  _Float16* Osq = qsm + C_ * OLD;                    // [8 ch][OLD]
  _Float16* Osk = qsm + C_ * OLD + 16 * OLD;

  const int tid = threadIdx.x, lane = tid & 31, m = tid >> 5;
  const int blk = blockIdx.x;
  const int b   = blk >> 7;
  const int rem = blk & 127;
  const int h0  = (rem >> 3) * 8;    // 16 h-tiles of 8 rows
  const int w0  = (rem & 7) * 16;    // 8 w-tiles of 16 cols

  // stage A = concat(x, y) channels as f16: As[s][c], s = hloc*16 + wloc
  for (int idx = tid; idx < 128 * 128; idx += 256) {
    int c = idx >> 7, s = idx & 127;
    int h = h0 + (s >> 4), w = w0 + (s & 15);
    float v = (c < C_) ? x[((b * C_ + c) * H_ + h) * W_ + w]
                       : y[((b * C_ + (c - C_)) * H_ + h) * W_ + w];
    As[s * ALD + c] = (_Float16)v;
  }
  // stage weights (f16, [och][ich] == B^T layout)
  for (int idx = tid; idx < C_ * 2 * C_; idx += 256) {
    int o = idx >> 7, c = idx & 127;
    Wv_h[o * WLD + c] = (_Float16)Wv[idx];
  }
  for (int idx = tid; idx < 16 * C_; idx += 256) {
    int o = idx >> 6, c = idx & 63;
    Wq_h[o * QWLD + c] = (o < C8_) ? (_Float16)Wq[o * C_ + c] : (_Float16)0.0f;
    Wk_h[o * QWLD + c] = (o < C8_) ? (_Float16)Wk[o * C_ + c] : (_Float16)0.0f;
  }
  __syncthreads();

  // v: 4 och-tiles x 4 K-chunks
  v8f ov[4];
#pragma unroll
  for (int n = 0; n < 4; ++n) ov[n] = vzero8();
#pragma unroll
  for (int kk = 0; kk < 4; ++kk) {
    v16h a = load_fragA(As, ALD, 16 * m, 32 * kk, lane);
#pragma unroll
    for (int n = 0; n < 4; ++n)
      ov[n] = wmma_f16f32(a, load_fragBT(Wv_h, WLD, 16 * n, 32 * kk, lane), ov[n]);
  }
  // q over x channels (K=64), k over y channels (K=64); och padded 8->16
  v8f oq = vzero8(), ok = vzero8();
#pragma unroll
  for (int kk = 0; kk < 2; ++kk) {
    v16h ax = load_fragA(As, ALD, 16 * m, 32 * kk, lane);
    oq = wmma_f16f32(ax, load_fragBT(Wq_h, QWLD, 0, 32 * kk, lane), oq);
    v16h ay = load_fragA(As, ALD, 16 * m, C_ + 32 * kk, lane);
    ok = wmma_f16f32(ay, load_fragBT(Wk_h, QWLD, 0, 32 * kk, lane), ok);
  }

  __syncthreads();   // all waves done reading As (aliased by Os*)

  // stage outputs (+bias) to LDS for layout-friendly global stores
#pragma unroll
  for (int g = 0; g < 8; ++g) {
    int s  = 16 * m + g + ((lane & 16) ? 8 : 0);
    int ch = lane & 15;
    if (ch < C8_) {
      Osq[ch * OLD + s] = (_Float16)(oq[g] + bq[ch]);
      Osk[ch * OLD + s] = (_Float16)(ok[g] + bk[ch]);
    }
#pragma unroll
    for (int n = 0; n < 4; ++n) {
      int chv = 16 * n + (lane & 15);
      Osv[chv * OLD + s] = (_Float16)(ov[n][g] + bv[chv]);
    }
  }
  __syncthreads();

  // global stores: [h][w] layout (lanes along w), [w][h] layout (lanes along h)
  for (int idx = tid; idx < C_ * 128; idx += 256) {
    int ch = idx >> 7, s = idx & 127;
    int h = h0 + (s >> 4), w = w0 + (s & 15);
    vbuf[((b * C_ + ch) * H_ + h) * W_ + w] = Osv[ch * OLD + s];
  }
  for (int idx = tid; idx < C_ * 128; idx += 256) {
    int ch = idx >> 7, t = idx & 127;
    int w = w0 + (t >> 3), h = h0 + (t & 7);
    int s = (t & 7) * 16 + (t >> 3);
    vT[((b * C_ + ch) * W_ + w) * H_ + h] = Osv[ch * OLD + s];
  }
  for (int idx = tid; idx < C8_ * 128; idx += 256) {
    int ch = idx >> 7, s = idx & 127;
    int h = h0 + (s >> 4), w = w0 + (s & 15);
    qbuf[((b * C8_ + ch) * H_ + h) * W_ + w] = Osq[ch * OLD + s];
    kbuf[((b * C8_ + ch) * H_ + h) * W_ + w] = Osk[ch * OLD + s];
  }
  for (int idx = tid; idx < C8_ * 128; idx += 256) {
    int ch = idx >> 7, t = idx & 127;
    int w = w0 + (t >> 3), h = h0 + (t & 7);
    int s = (t & 7) * 16 + (t >> 3);
    qT[((b * C8_ + ch) * W_ + w) * H_ + h] = Osq[ch * OLD + s];
    kT[((b * C8_ + ch) * W_ + w) * H_ + h] = Osk[ch * OLD + s];
  }
}

// ---------------------------------------------------------------------------
// Kernel 2: row attention. One workgroup per (b, row r). 8 waves, wave m owns
// query tile rows 16m..16m+15. Writes partial softmax stats + O_row to ws.
// ---------------------------------------------------------------------------
__global__ __launch_bounds__(256)
void cc_row_kernel(const _Float16* __restrict__ qbuf,
                   const _Float16* __restrict__ kbuf,
                   const _Float16* __restrict__ vbuf,
                   float* __restrict__ m_ws, float* __restrict__ s_ws,
                   float* __restrict__ o_ws)
{
  __shared__ alignas(16) _Float16 smem[SMEM_HALVES];
  _Float16* Qs = smem;                    // [128][QK_LD], K padded to 32
  _Float16* Ks = smem + W_ * QK_LD;
  _Float16* Ps = smem;                    // aliases Q/K region after barrier
  _Float16* Vs = smem + PS_ELEMS;         // [64][VS_LD]

  const int br = blockIdx.x;
  const int b = br / H_, r = br % H_;
  const int tid = threadIdx.x, lane = tid & 31, m = tid >> 5;

  for (int idx = tid; idx < W_ * 32; idx += 256) {
    int kk = idx / W_, i = idx % W_;      // i = column position
    _Float16 qv = (_Float16)0.0f, kv = (_Float16)0.0f;
    if (kk < C8_) {
      qv = qbuf[((b * C8_ + kk) * H_ + r) * W_ + i];
      kv = kbuf[((b * C8_ + kk) * H_ + r) * W_ + i];
    }
    Qs[i * QK_LD + kk] = qv;
    Ks[i * QK_LD + kk] = kv;
  }
  for (int idx = tid; idx < C_ * W_; idx += 256) {
    int ch = idx / W_, j = idx % W_;
    Vs[ch * VS_LD + j] = vbuf[((b * C_ + ch) * H_ + r) * W_ + j];
  }
  __syncthreads();

  // S = Q_row (128x8, padded to 32) * K_row^T
  v16h aq = load_fragA(Qs, QK_LD, 16 * m, 0, lane);
  v8f acc[8];
#pragma unroll
  for (int n = 0; n < 8; ++n) {
    v16h bk = load_fragBT(Ks, QK_LD, 16 * n, 0, lane);
    acc[n] = wmma_f16f32(aq, bk, vzero8());
  }

  // per-query (row-part) max & sum-exp; overwrite acc with exp(S - m)
  float mrow[8], srow[8];
#pragma unroll
  for (int g = 0; g < 8; ++g) {
    float mv = acc[0][g];
#pragma unroll
    for (int n = 1; n < 8; ++n) mv = fmaxf(mv, acc[n][g]);
#pragma unroll
    for (int off = 1; off <= 8; off <<= 1) mv = fmaxf(mv, __shfl_xor(mv, off, 32));
    float sv = 0.0f;
#pragma unroll
    for (int n = 0; n < 8; ++n) {
      float e = __expf(acc[n][g] - mv);
      acc[n][g] = e;
      sv += e;
    }
#pragma unroll
    for (int off = 1; off <= 8; off <<= 1) sv += __shfl_xor(sv, off, 32);
    mrow[g] = mv; srow[g] = sv;
  }

  __syncthreads();  // everyone done reading Qs/Ks (region aliased by Ps)

#pragma unroll
  for (int n = 0; n < 8; ++n)
#pragma unroll
    for (int g = 0; g < 8; ++g) {
      int i = 16 * m + g + ((lane & 16) ? 8 : 0);
      int j = 16 * n + (lane & 15);
      Ps[i * PS_LD + j] = (_Float16)acc[n][g];
    }

  if ((lane & 15) == 0) {
#pragma unroll
    for (int g = 0; g < 8; ++g) {
      int i = 16 * m + g + ((lane & 16) ? 8 : 0);
      m_ws[(b * H_ + r) * W_ + i] = mrow[g];
      s_ws[(b * H_ + r) * W_ + i] = srow[g];
    }
  }

  // O_row = P (128x128) * V_row^T (128x64)
  v8f o[4];
#pragma unroll
  for (int n = 0; n < 4; ++n) o[n] = vzero8();
#pragma unroll
  for (int kk = 0; kk < 4; ++kk) {
    v16h ap = load_fragA(Ps, PS_LD, 16 * m, 32 * kk, lane);
#pragma unroll
    for (int n = 0; n < 4; ++n) {
      v16h bv = load_fragBT(Vs, VS_LD, 16 * n, 32 * kk, lane);
      o[n] = wmma_f16f32(ap, bv, o[n]);
    }
  }

#pragma unroll
  for (int n = 0; n < 4; ++n)
#pragma unroll
    for (int g = 0; g < 8; ++g) {
      int i  = 16 * m + g + ((lane & 16) ? 8 : 0);
      int ch = 16 * n + (lane & 15);
      o_ws[((b * H_ + r) * W_ + i) * C_ + ch] = o[n][g];
    }
}

// ---------------------------------------------------------------------------
// Kernel 3: column attention (diag masked) + joint-softmax merge + residual.
// One workgroup per (b, column w).
// ---------------------------------------------------------------------------
__global__ __launch_bounds__(256)
void cc_col_kernel(const _Float16* __restrict__ qT,
                   const _Float16* __restrict__ kT,
                   const _Float16* __restrict__ vT,
                   const float* __restrict__ x, const float* __restrict__ y,
                   const float* __restrict__ m_ws, const float* __restrict__ s_ws,
                   const float* __restrict__ o_ws, const float* __restrict__ gptr,
                   float* __restrict__ out)
{
  __shared__ alignas(16) _Float16 smem[SMEM_HALVES];
  _Float16* Qs = smem;
  _Float16* Ks = smem + H_ * QK_LD;
  _Float16* Ps = smem;
  _Float16* Vs = smem + PS_ELEMS;

  const int bw = blockIdx.x;
  const int b = bw / W_, w = bw % W_;
  const int tid = threadIdx.x, lane = tid & 31, m = tid >> 5;

  for (int idx = tid; idx < H_ * 32; idx += 256) {
    int kk = idx / H_, i = idx % H_;      // i = row position
    _Float16 qv = (_Float16)0.0f, kv = (_Float16)0.0f;
    if (kk < C8_) {
      qv = qT[((b * C8_ + kk) * W_ + w) * H_ + i];
      kv = kT[((b * C8_ + kk) * W_ + w) * H_ + i];
    }
    Qs[i * QK_LD + kk] = qv;
    Ks[i * QK_LD + kk] = kv;
  }
  for (int idx = tid; idx < C_ * H_; idx += 256) {
    int ch = idx / H_, j = idx % H_;
    Vs[ch * VS_LD + j] = vT[((b * C_ + ch) * W_ + w) * H_ + j];
  }
  __syncthreads();

  v16h aq = load_fragA(Qs, QK_LD, 16 * m, 0, lane);
  v8f acc[8];
#pragma unroll
  for (int n = 0; n < 8; ++n) {
    v16h bk = load_fragBT(Ks, QK_LD, 16 * n, 0, lane);
    acc[n] = wmma_f16f32(aq, bk, vzero8());
  }

  // mask the i==j diagonal (only the n==m tile can contain it)
#pragma unroll
  for (int n = 0; n < 8; ++n)
    if (n == m) {
#pragma unroll
      for (int g = 0; g < 8; ++g)
        if ((lane & 15) == g + ((lane & 16) ? 8 : 0)) acc[n][g] = -1e30f;
    }

  float mrow[8], srow[8];
#pragma unroll
  for (int g = 0; g < 8; ++g) {
    float mv = acc[0][g];
#pragma unroll
    for (int n = 1; n < 8; ++n) mv = fmaxf(mv, acc[n][g]);
#pragma unroll
    for (int off = 1; off <= 8; off <<= 1) mv = fmaxf(mv, __shfl_xor(mv, off, 32));
    float sv = 0.0f;
#pragma unroll
    for (int n = 0; n < 8; ++n) {
      float e = __expf(acc[n][g] - mv);
      acc[n][g] = e;
      sv += e;
    }
#pragma unroll
    for (int off = 1; off <= 8; off <<= 1) sv += __shfl_xor(sv, off, 32);
    mrow[g] = mv; srow[g] = sv;
  }

  __syncthreads();

#pragma unroll
  for (int n = 0; n < 8; ++n)
#pragma unroll
    for (int g = 0; g < 8; ++g) {
      int i = 16 * m + g + ((lane & 16) ? 8 : 0);
      int j = 16 * n + (lane & 15);
      Ps[i * PS_LD + j] = (_Float16)acc[n][g];
    }

  v8f o[4];
#pragma unroll
  for (int n = 0; n < 4; ++n) o[n] = vzero8();
#pragma unroll
  for (int kk = 0; kk < 4; ++kk) {
    v16h ap = load_fragA(Ps, PS_LD, 16 * m, 32 * kk, lane);
#pragma unroll
    for (int n = 0; n < 4; ++n) {
      v16h bv = load_fragBT(Vs, VS_LD, 16 * n, 32 * kk, lane);
      o[n] = wmma_f16f32(ap, bv, o[n]);
    }
  }

  // merge column part with stored row part: joint softmax normalization
  const float gamma = *gptr;
#pragma unroll
  for (int n = 0; n < 4; ++n)
#pragma unroll
    for (int g = 0; g < 8; ++g) {
      int i  = 16 * m + g + ((lane & 16) ? 8 : 0);   // pixel row
      int ch = 16 * n + (lane & 15);
      float mc = mrow[g], sc = srow[g];
      int pidx = (b * H_ + i) * W_ + w;
      float mr  = m_ws[pidx];
      float sr  = s_ws[pidx];
      float orv = o_ws[pidx * C_ + ch];
      float mm  = fmaxf(mr, mc);
      float er  = __expf(mr - mm);
      float ec  = __expf(mc - mm);
      float inv = 1.0f / (sr * er + sc * ec);
      int gi = ((b * C_ + ch) * H_ + i) * W_ + w;
      out[gi] = gamma * ((orv * er + o[n][g] * ec) * inv) + x[gi] + y[gi];
    }
}

// ---------------------------------------------------------------------------
extern "C" void kernel_launch(void* const* d_in, const int* in_sizes, int n_in,
                              void* d_out, int out_size, void* d_ws, size_t ws_size,
                              hipStream_t stream)
{
  (void)in_sizes; (void)n_in; (void)out_size; (void)ws_size;
  const float* x    = (const float*)d_in[0];
  const float* y    = (const float*)d_in[1];
  const float* Wq   = (const float*)d_in[2];
  const float* bq   = (const float*)d_in[3];
  const float* Wk   = (const float*)d_in[4];
  const float* bk   = (const float*)d_in[5];
  const float* Wv   = (const float*)d_in[6];
  const float* bv   = (const float*)d_in[7];
  const float* gptr = (const float*)d_in[8];
  float* out = (float*)d_out;

  char* ws = (char*)d_ws;
  size_t off = 0;
  auto take = [&](size_t bytes) -> char* {
    char* p = ws + off;
    off += (bytes + 255) & ~(size_t)255;
    return p;
  };
  const size_t nq = (size_t)B_ * C8_ * H_ * W_;
  const size_t nv = (size_t)B_ * C_ * H_ * W_;
  const size_t np = (size_t)B_ * H_ * W_;

  _Float16* qbuf = (_Float16*)take(nq * 2);
  _Float16* kbuf = (_Float16*)take(nq * 2);
  _Float16* qT   = (_Float16*)take(nq * 2);
  _Float16* kT   = (_Float16*)take(nq * 2);
  _Float16* vbuf = (_Float16*)take(nv * 2);
  _Float16* vT   = (_Float16*)take(nv * 2);
  float* m_ws = (float*)take(np * 4);
  float* s_ws = (float*)take(np * 4);
  float* o_ws = (float*)take(np * (size_t)C_ * 4);

  cc_qkv_kernel<<<B_ * (H_ / 8) * (W_ / 16), 256, 0, stream>>>(
      x, y, Wq, bq, Wk, bk, Wv, bv, qbuf, kbuf, qT, kT, vbuf, vT);
  cc_row_kernel<<<B_ * H_, 256, 0, stream>>>(qbuf, kbuf, vbuf, m_ws, s_ws, o_ws);
  cc_col_kernel<<<B_ * W_, 256, 0, stream>>>(qT, kT, vT, x, y, m_ws, s_ws, o_ws,
                                             gptr, out);
}
